// CrossAttention_86088324481795
// MI455X (gfx1250) — matrix operand order repaired and checked by reading, not compiled, not run on hardware
//
#include <hip/hip_runtime.h>
#include <math.h>

#define BB 2
#define NQ 1024
#define NK 2048
#define DIMD 1024
#define HH 16
#define HD 64

typedef __attribute__((ext_vector_type(16))) __bf16 v16bf;
typedef __attribute__((ext_vector_type(8)))  __bf16 v8bf;
typedef __attribute__((ext_vector_type(8)))  float  v8f;

union V16U { v16bf v; v8bf h[2]; };

static __device__ __forceinline__ v8f wmma_bf16(v16bf a, v16bf b, v8f c) {
  // D = A(16x32 bf16) x B(32x16 bf16) + C(16x16 f32)
  return __builtin_amdgcn_wmma_f32_16x16x32_bf16(false, a, false, b, (short)0, c,
                                                 false, false);
}

// DPP-based max-combine within 16-lane rows (pure VALU, no LDS round trip).
template <int CTRL>
static __device__ __forceinline__ float dpp_max(float x) {
  int s = __builtin_amdgcn_update_dpp(0, __float_as_int(x), CTRL, 0xF, 0xF, true);
  return fmaxf(x, __int_as_float(s));
}
// full 16-lane butterfly: xor1, xor2, half-mirror(8), mirror(16)
static __device__ __forceinline__ float rowmax16(float x) {
  x = dpp_max<0xB1>(x);   // quad_perm [1,0,3,2]
  x = dpp_max<0x4E>(x);   // quad_perm [2,3,0,1]
  x = dpp_max<0x141>(x);  // ROW_HALF_MIRROR
  x = dpp_max<0x140>(x);  // ROW_MIRROR
  return x;
}

// ---------------------------------------------------------------------------
// Pass 1: masked batch sum of distances  -> stats[0..1];  stats[2..3] = coef_b
// ---------------------------------------------------------------------------
__global__ void init_stats(float* stats) {
  if (threadIdx.x < 4) stats[threadIdx.x] = 0.f;
}

__global__ void dist_reduce(const float* __restrict__ dist,
                            const unsigned char* __restrict__ am,
                            const unsigned char* __restrict__ kpmk,
                            float* __restrict__ stats) {
  const long total = (long)BB * NQ * NK;
  float s0 = 0.f, s1 = 0.f;
  for (long i = (long)blockIdx.x * blockDim.x + threadIdx.x; i < total;
       i += (long)gridDim.x * blockDim.x) {
    const int b = (int)(i / ((long)NQ * NK));
    const int kcol = (int)(i % NK);
    if (am[i] && kpmk[b * NK + kcol]) {
      if (b == 0) s0 += dist[i]; else s1 += dist[i];
    }
  }
  for (int off = 16; off; off >>= 1) {
    s0 += __shfl_xor(s0, off, 32);
    s1 += __shfl_xor(s1, off, 32);
  }
  if ((threadIdx.x & 31) == 0) {
    atomicAdd(&stats[0], s0);
    atomicAdd(&stats[1], s1);
  }
}

__global__ void compute_coef(const float* __restrict__ ga, float* __restrict__ stats) {
  if (threadIdx.x == 0 && blockIdx.x == 0) {
    const float alpha = logf(1.f + expf(ga[0]));  // softplus
    for (int b = 0; b < BB; ++b) {
      float mean = stats[b] / ((float)NQ * (float)NK + 1e-6f);
      mean = fmaxf(mean, 1e-6f);
      stats[2 + b] = alpha / mean;
    }
  }
}

// ---------------------------------------------------------------------------
// WMMA GEMM: C(M,N) = A(M,K) * W(N,K)^T ; A,W fp32 in memory, bf16 in LDS.
// Block tile 32(M) x 64(N) x 32(K); 8 waves, each computes one 16x16 tile.
// ---------------------------------------------------------------------------
template <bool STORE_BF16>
__global__ __launch_bounds__(256) void gemm_wmma(const float* __restrict__ A,
                                                 const float* __restrict__ W,
                                                 void* __restrict__ Cout,
                                                 int M, int Ncols, int K) {
  __shared__ __align__(64) __bf16 As[32][32];
  __shared__ __align__(64) __bf16 Ws[64][32];
  const int tid = threadIdx.x;
  const int lane = tid & 31, wave = tid >> 5;
  const int ln = lane & 15, hl = lane >> 4;
  const int m0 = blockIdx.y * 32;
  const int n0 = blockIdx.x * 64;
  const int wm = wave >> 2;  // 0..1  (row tile)
  const int wn = wave & 3;   // 0..3  (col tile)
  v8f acc = {};

  for (int kk = 0; kk < K; kk += 32) {
    __syncthreads();
    {  // A tile: 32x32 fp32 -> bf16, 4 elems/thread
      const int idx = tid * 4;
      const int r = idx >> 5, c = idx & 31;
      const float4 f = *(const float4*)&A[(long)(m0 + r) * K + kk + c];
      As[r][c + 0] = (__bf16)f.x; As[r][c + 1] = (__bf16)f.y;
      As[r][c + 2] = (__bf16)f.z; As[r][c + 3] = (__bf16)f.w;
    }
    {  // W tile: 64x32 fp32 -> bf16, 8 elems/thread
      const int idx = tid * 8;
      const int r = idx >> 5, c = idx & 31;
      const float4 f0 = *(const float4*)&W[(long)(n0 + r) * K + kk + c];
      const float4 f1 = *(const float4*)&W[(long)(n0 + r) * K + kk + c + 4];
      v8bf wv;
      wv[0] = (__bf16)f0.x; wv[1] = (__bf16)f0.y; wv[2] = (__bf16)f0.z; wv[3] = (__bf16)f0.w;
      wv[4] = (__bf16)f1.x; wv[5] = (__bf16)f1.y; wv[6] = (__bf16)f1.z; wv[7] = (__bf16)f1.w;
      *(v8bf*)&Ws[r][c] = wv;
    }
    __syncthreads();

    V16U af, wf;
    // A fragment: row = ln; half 0 holds K 0..15, half 1 K 16..31
    af.h[0] = *(const v8bf*)&As[wm * 16 + ln][hl * 16];
    af.h[1] = *(const v8bf*)&As[wm * 16 + ln][hl * 16 + 8];
    // B fragment: col = ln; half 0 K=0..15, half 1 K=16..31 (contiguous in row)
    wf.v = *(const v16bf*)&Ws[wn * 16 + ln][hl * 16];
    acc = wmma_bf16(af.v, wf.v, acc);
  }

  for (int r = 0; r < 8; ++r) {  // C layout: VGPR r = row r (+8 for upper half)
    const int row = m0 + wm * 16 + r + hl * 8;
    const int col = n0 + wn * 16 + ln;
    if (STORE_BF16) ((__bf16*)Cout)[(long)row * Ncols + col] = (__bf16)acc[r];
    else            ((float*)Cout)[(long)row * Ncols + col] = acc[r];
  }
}

// ---------------------------------------------------------------------------
// Fused flash attention with distance gating.
//   w = softmax( qp.kp^T/sqrt(HD) - coef_b * distances ; masked -> -1e30 )
//   ctx = (w @ vp) * kpm_q
// Block = 8 waves, one (b,h), 128 q rows (16 per wave); 32 keys per step.
// K tiles: double-buffered async global->LDS copies (ASYNCcnt pipelining).
// V tiles: double-buffered register prefetch + LDS transpose.
// ---------------------------------------------------------------------------
__global__ __launch_bounds__(256) void flash_gate(
    const __bf16* __restrict__ qp, const __bf16* __restrict__ kp,
    const __bf16* __restrict__ vp, const float* __restrict__ dist,
    const unsigned char* __restrict__ am, const unsigned char* __restrict__ kpmq,
    const unsigned char* __restrict__ kpmk, const float* __restrict__ stats,
    float* __restrict__ ctx) {
  __shared__ __align__(64) __bf16 Kt[2][32][64];    // keys x dims, double buffered
  __shared__ __align__(64) __bf16 Vt[2][64][32];    // dims x keys (transposed), x2
  __shared__ __align__(64) __bf16 Pw[8][16][32];    // per-wave P scratch
  const int tid = threadIdx.x, lane = tid & 31, wave = tid >> 5;
  const int ln = lane & 15, hl = lane >> 4;
  const int h = blockIdx.y, b = blockIdx.z;
  const int q0 = blockIdx.x * 128 + wave * 16;
  const float coefb = stats[2 + b];

  // Q fragments kept in registers for the whole key loop (A-matrix layout)
  V16U qf[2];
  {
    const long qbase = ((long)(b * NQ + q0 + ln)) * (HH * HD) + h * HD;
    for (int kki = 0; kki < 2; ++kki) {
      const __bf16* p = qp + qbase + kki * 32 + hl * 16;
      qf[kki].h[0] = *(const v8bf*)p;
      qf[kki].h[1] = *(const v8bf*)(p + 8);
    }
  }

  // all-ones B fragment: rowsum(P) = P @ ones via one extra WMMA
  v16bf ones;
  #pragma unroll
  for (int i = 0; i < 16; ++i) ones[i] = (__bf16)1.0f;

  // per-thread cooperative-load geometry (8 halves = 16B per thread)
  const int cidx = tid * 8;
  const int ckey = cidx >> 6, cd0 = cidx & 63;
  unsigned kofs[2];
  kofs[0] = (unsigned)(unsigned long long)
      (__attribute__((address_space(3))) void*)&Kt[0][ckey][cd0];
  kofs[1] = (unsigned)(unsigned long long)
      (__attribute__((address_space(3))) void*)&Kt[1][ckey][cd0];
  const long tilebase = ((long)(b * NK + ckey)) * (HH * HD) + h * HD + cd0;

  float mrow[8], lrow[8], corr[8];
  v8f o0 = {}, o1 = {}, o2 = {}, o3 = {};
  for (int r = 0; r < 8; ++r) { mrow[r] = -3e38f; lrow[r] = 0.f; }

  // prologue: start async K(0) and register-prefetch V(0)
  {
    const unsigned long long ka = (unsigned long long)(kp + tilebase);
    asm volatile("global_load_async_to_lds_b128 %0, %1, off"
                 :: "v"(kofs[0]), "v"(ka) : "memory");
  }
  v8bf vv = *(const v8bf*)(vp + tilebase);

  const int NT = NK / 32;
  for (int kt = 0; kt < NT; ++kt) {
    const int bsel = kt & 1;
    // retire all readers of the buffers we are about to overwrite
    __syncthreads();

    // scatter prefetched V(kt) registers into transposed LDS tile
    #pragma unroll
    for (int j = 0; j < 8; ++j) Vt[bsel][cd0 + j][ckey] = vv[j];

    v8bf vvn = vv;
    const bool more = (kt + 1 < NT);
    if (more) {  // start async K(kt+1) into the other buffer; prefetch V(kt+1)
      const long nb = tilebase + (long)(kt + 1) * 32 * (HH * HD);
      const unsigned long long ka = (unsigned long long)(kp + nb);
      asm volatile("global_load_async_to_lds_b128 %0, %1, off"
                   :: "v"(kofs[bsel ^ 1]), "v"(ka) : "memory");
      vvn = *(const v8bf*)(vp + nb);
    }
    // async copies complete in order: cnt<=1 means tile kt has landed
    if (more) asm volatile("s_wait_asynccnt 1" ::: "memory");
    else      asm volatile("s_wait_asynccnt 0" ::: "memory");
    __syncthreads();

    const int k0 = kt * 32;

    // S(16x32) = Q(16x64) . K_tile(32x64)^T : 4 WMMAs
    v8f s0 = {}, s1 = {};
    for (int kki = 0; kki < 2; ++kki) {
      v16bf bk0 = *(const v16bf*)&Kt[bsel][ln][kki * 32 + hl * 16];
      v16bf bk1 = *(const v16bf*)&Kt[bsel][16 + ln][kki * 32 + hl * 16];
      s0 = wmma_bf16(qf[kki].v, bk0, s0);
      s1 = wmma_bf16(qf[kki].v, bk1, s1);
    }

    // mask + distance gate in logit space; running max via DPP butterflies
    const int col0 = k0 + ln, col1 = k0 + 16 + ln;
    const unsigned char kb0 = kpmk[b * NK + col0];
    const unsigned char kb1 = kpmk[b * NK + col1];
    for (int r = 0; r < 8; ++r) {
      const int qrow = q0 + r + hl * 8;
      const long ri = ((long)(b * NQ + qrow)) * NK;
      const float x0 = (am[ri + col0] && kb0)
                           ? (s0[r] * 0.125f - coefb * dist[ri + col0]) : -1e30f;
      const float x1 = (am[ri + col1] && kb1)
                           ? (s1[r] * 0.125f - coefb * dist[ri + col1]) : -1e30f;
      const float mt = rowmax16(fmaxf(x0, x1));
      const float mnew = fmaxf(mrow[r], mt);
      corr[r] = __expf(mrow[r] - mnew);
      mrow[r] = mnew;
      const float p0 = __expf(x0 - mnew);
      const float p1 = __expf(x1 - mnew);
      // C-layout -> LDS (row r / r+8 per half, col = ln / 16+ln)
      Pw[wave][r + hl * 8][ln] = (__bf16)p0;
      Pw[wave][r + hl * 8][16 + ln] = (__bf16)p1;
    }
    for (int r = 0; r < 8; ++r) {
      o0[r] *= corr[r]; o1[r] *= corr[r]; o2[r] *= corr[r]; o3[r] *= corr[r];
    }

    // per-wave LDS round trip: DS ops are in-order within a wave; wait + fence
    asm volatile("s_wait_dscnt 0" ::: "memory");
    __builtin_amdgcn_wave_barrier();

    // P as A-matrix fragment (16x32)
    V16U pf;
    pf.h[0] = *(const v8bf*)&Pw[wave][ln][hl * 16];
    pf.h[1] = *(const v8bf*)&Pw[wave][ln][hl * 16 + 8];

    // row sums of P via WMMA against all-ones (replicated across columns)
    v8f zero = {};
    v8f rsv = wmma_bf16(pf.v, ones, zero);
    for (int r = 0; r < 8; ++r) lrow[r] = lrow[r] * corr[r] + rsv[r];

    // V^T as B-matrix fragments: col = dim, K = keys (contiguous in Vt rows)
    v16bf bv0 = *(const v16bf*)&Vt[bsel][ 0 + ln][hl * 16];
    v16bf bv1 = *(const v16bf*)&Vt[bsel][16 + ln][hl * 16];
    v16bf bv2 = *(const v16bf*)&Vt[bsel][32 + ln][hl * 16];
    v16bf bv3 = *(const v16bf*)&Vt[bsel][48 + ln][hl * 16];
    o0 = wmma_bf16(pf.v, bv0, o0);
    o1 = wmma_bf16(pf.v, bv1, o1);
    o2 = wmma_bf16(pf.v, bv2, o2);
    o3 = wmma_bf16(pf.v, bv3, o3);

    vv = vvn;
  }

  for (int r = 0; r < 8; ++r) {
    const int qrow = q0 + r + hl * 8;
    const float f = kpmq[b * NQ + qrow] ? (1.f / (lrow[r] + 1e-6f)) : 0.f;
    float* dst = ctx + ((long)(b * NQ + qrow)) * (HH * HD) + h * HD;
    dst[ 0 + ln] = o0[r] * f;
    dst[16 + ln] = o1[r] * f;
    dst[32 + ln] = o2[r] * f;
    dst[48 + ln] = o3[r] * f;
  }
}

// ---------------------------------------------------------------------------
extern "C" void kernel_launch(void* const* d_in, const int* in_sizes, int n_in,
                              void* d_out, int out_size, void* d_ws, size_t ws_size,
                              hipStream_t stream) {
  (void)in_sizes; (void)n_in; (void)out_size; (void)ws_size;
  const float* q    = (const float*)d_in[0];
  const float* k    = (const float*)d_in[1];
  const float* v    = (const float*)d_in[2];
  const float* dist = (const float*)d_in[3];
  const unsigned char* am   = (const unsigned char*)d_in[4];
  const unsigned char* kpmq = (const unsigned char*)d_in[5];
  const unsigned char* kpmk = (const unsigned char*)d_in[6];
  const float* Wq = (const float*)d_in[7];
  const float* Wk = (const float*)d_in[8];
  const float* Wv = (const float*)d_in[9];
  const float* Wo = (const float*)d_in[10];
  const float* ga = (const float*)d_in[11];

  char* ws = (char*)d_ws;
  __bf16* qp  = (__bf16*)(ws);                                // 2048*1024 bf16 = 4MB
  __bf16* kp  = (__bf16*)(ws + (size_t)4  * 1024 * 1024);     // 4096*1024 bf16 = 8MB
  __bf16* vp  = (__bf16*)(ws + (size_t)12 * 1024 * 1024);     // 8MB
  float*  ctx = (float*)(ws + (size_t)20 * 1024 * 1024);      // 2048*1024 f32 = 8MB
  float*  stats = (float*)(ws + (size_t)28 * 1024 * 1024);    // 4 floats

  init_stats<<<1, 32, 0, stream>>>(stats);
  dist_reduce<<<512, 256, 0, stream>>>(dist, am, kpmk, stats);
  compute_coef<<<1, 32, 0, stream>>>(ga, stats);

  dim3 blk(256);
  // projections: X(M,1024) @ W(1024,1024)^T -> bf16
  gemm_wmma<true><<<dim3(16, 64),  blk, 0, stream>>>(q, Wq, qp, BB * NQ, HH * HD, DIMD);
  gemm_wmma<true><<<dim3(16, 128), blk, 0, stream>>>(k, Wk, kp, BB * NK, HH * HD, DIMD);
  gemm_wmma<true><<<dim3(16, 128), blk, 0, stream>>>(v, Wv, vp, BB * NK, HH * HD, DIMD);

  // fused gated attention: grid (q-blocks, heads, batch)
  flash_gate<<<dim3(NQ / 128, HH, BB), blk, 0, stream>>>(qp, kp, vp, dist, am,
                                                         kpmq, kpmk, stats, ctx);

  // output projection: ctx(2048,1024) @ Wo(1024,1024)^T -> f32 d_out
  gemm_wmma<false><<<dim3(16, 64), blk, 0, stream>>>(ctx, Wo, (float*)d_out,
                                                     BB * NQ, DIMD, HH * HD);
}
